// Model_58755152609875
// MI455X (gfx1250) — matrix operand order repaired and checked by reading, not compiled, not run on hardware
//
#include <hip/hip_runtime.h>
#include <math.h>

// ---------------------------------------------------------------------------
// Problem constants (from the reference)
// ---------------------------------------------------------------------------
#define N_B    4
#define M_SEQ  2048
#define DIM    512
#define HEADS  8
#define HDIM   64          // DIM / HEADS
#define TOPK   16
#define BH     (N_B * HEADS)      // 32 batch-heads
#define QKV_N  (3 * DIM)          // 1536
#define ROWS   (N_B * M_SEQ)      // 8192
#define SCALE  0.04419417382415922f   // 512^-0.5

typedef __attribute__((ext_vector_type(16))) _Float16 v16h_t;
typedef __attribute__((ext_vector_type(8)))  _Float16 v8h_t;
typedef __attribute__((ext_vector_type(8)))  float    v8f_t;
typedef __attribute__((ext_vector_type(2)))  int      v2i_t;   // b64 payload

#if defined(__gfx1250__) && __has_builtin(__builtin_amdgcn_global_load_async_to_lds_b64)
#define HAS_ASYNC_LDS 1
#else
#define HAS_ASYNC_LDS 0
#endif

__device__ __forceinline__ int lane_id() { return threadIdx.x & 31; }

__device__ __forceinline__ void wait_async_zero() {
#if HAS_ASYNC_LDS
#if __has_builtin(__builtin_amdgcn_s_wait_asynccnt)
  __builtin_amdgcn_s_wait_asynccnt(0);
#else
  asm volatile("s_wait_asynccnt 0x0" ::: "memory");
#endif
#endif
}

// ---------------------------------------------------------------------------
// WMMA fragment loaders. Layouts used by this code (wave32; contiguous k, not the ISA table order):
//  A 16x32 f16 : lane L -> M = L&15 ; K = k0 + (L>=16 ? 16:0) + e (contiguous)
//  B 32x16 f16 : lane L -> N = L&15 ; K = k0 + (L>=16 ? 16:0) + e (contiguous)
//  C 16x16 f32 : elem r -> M = r + (L>=16 ? 8:0), N = L&15
// ---------------------------------------------------------------------------
__device__ __forceinline__ v16h_t load_frag_a(const _Float16* __restrict__ A,
                                              int lda, int row0, int k0) {
  const int l = lane_id();
  const _Float16* p = A + (size_t)(row0 + (l & 15)) * lda + k0 + ((l >> 4) << 4);
  union { v16h_t f; v8h_t h[2]; } u;
  u.h[0] = *(const v8h_t*)(p);
  u.h[1] = *(const v8h_t*)(p + 8);
  return u.f;
}

__device__ __forceinline__ v16h_t load_frag_b(const _Float16* __restrict__ B,
                                              int ldb, int col0, int k0) {
  const int l = lane_id();
  const _Float16* p = B + (size_t)(col0 + (l & 15)) * ldb + k0 + ((l >> 4) << 4);
  union { v16h_t f; v8h_t h[2]; } u;
  u.h[0] = *(const v8h_t*)(p);
  u.h[1] = *(const v8h_t*)(p + 8);
  return u.f;
}

// B fragment out of an LDS-resident 16x64 key tile (row-major, ld = HDIM).
__device__ __forceinline__ v16h_t load_frag_b_lds(const _Float16* kls, int k0) {
  const int l = lane_id();
  const _Float16* p = kls + (l & 15) * HDIM + k0 + ((l >> 4) << 4);
  union { v16h_t f; v8h_t h[2]; } u;
  u.h[0] = *(const v8h_t*)(p);
  u.h[1] = *(const v8h_t*)(p + 8);
  return u.f;
}

__device__ __forceinline__ v8f_t wmma_f16(v16h_t a, v16h_t b, v8f_t c) {
  return __builtin_amdgcn_wmma_f32_16x16x32_f16(false, a, false, b,
                                                (short)0, c, false, false);
}

// Stage one 16x64 f16 key tile (2048 B) global -> LDS with 256 threads,
// 8 B per thread. Async DMA path on gfx1250 (tracked by ASYNCcnt).
__device__ __forceinline__ void stage_tile(const _Float16* __restrict__ kb,
                                           _Float16* kls, int key0) {
  const int t   = threadIdx.x;        // 0..255
  const int row = t >> 4;             // 0..15
  const int col = (t & 15) << 2;      // halves, 8B-aligned
  const _Float16* g = kb + (size_t)(key0 + row) * HDIM + col;
  _Float16* d = kls + row * HDIM + col;
#if HAS_ASYNC_LDS
  __builtin_amdgcn_global_load_async_to_lds_b64((v2i_t*)g, (v2i_t*)d, 0, 0);
#else
  *(unsigned long long*)d = *(const unsigned long long*)g;
#endif
}

// ---------------------------------------------------------------------------
// Kernel 1: depthwise conv1d(k=3, pad=1) over m + bias + residual  -> f16 x'
// ---------------------------------------------------------------------------
__global__ void conv_pos_kernel(const float* __restrict__ x,
                                const float* __restrict__ pos_w,
                                const float* __restrict__ pos_b,
                                _Float16* __restrict__ xh) {
  int i = blockIdx.x * blockDim.x + threadIdx.x;
  if (i >= N_B * M_SEQ * DIM) return;
  int c = i % DIM;
  int m = (i / DIM) % M_SEQ;
  int n = i / (DIM * M_SEQ);
  const float* xr = x + (size_t)n * M_SEQ * DIM + c;
  float w0 = pos_w[c * 3 + 0], w1 = pos_w[c * 3 + 1], w2 = pos_w[c * 3 + 2];
  float acc = xr[(size_t)m * DIM] * w1 + pos_b[c];
  if (m > 0)         acc += xr[(size_t)(m - 1) * DIM] * w0;
  if (m < M_SEQ - 1) acc += xr[(size_t)(m + 1) * DIM] * w2;
  xh[i] = (_Float16)(x[i] + acc);
}

// ---------------------------------------------------------------------------
// Kernel 2: f32 -> f16 weight conversion (qkv_w then wo_w)
// ---------------------------------------------------------------------------
__global__ void cvt_weights_kernel(const float* __restrict__ qkv_w,
                                   const float* __restrict__ wo_w,
                                   _Float16* __restrict__ wq_h,
                                   _Float16* __restrict__ wo_h) {
  int i = blockIdx.x * blockDim.x + threadIdx.x;
  const int n1 = QKV_N * DIM;
  if (i < n1)                    wq_h[i]      = (_Float16)qkv_w[i];
  else if (i < n1 + DIM * DIM)   wo_h[i - n1] = (_Float16)wo_w[i - n1];
}

// ---------------------------------------------------------------------------
// Kernel 3: QKV GEMM  C(8192x1536) = x'(8192x512) * Wqkv^T + b
// 8 waves; block tile 128x32; wave tile 16x32; head-major scatter epilogue.
// ---------------------------------------------------------------------------
__global__ __launch_bounds__(256)
void qkv_gemm_kernel(const _Float16* __restrict__ xh,
                     const _Float16* __restrict__ wq_h,
                     const float* __restrict__ qkv_b,
                     float* __restrict__ q_f,
                     _Float16* __restrict__ q_h,
                     _Float16* __restrict__ k_h,
                     _Float16* __restrict__ v_h) {
  const int w = threadIdx.x >> 5;
  const int l = lane_id();
  const int row0 = blockIdx.y * 128 + w * 16;
  const int col0 = blockIdx.x * 32;
  v8f_t acc0 = {}, acc1 = {};
  for (int k0 = 0; k0 < DIM; k0 += 32) {
    v16h_t a  = load_frag_a(xh, DIM, row0, k0);
    if (k0 + 32 < DIM)
      __builtin_prefetch(xh + (size_t)(row0 + (l & 15)) * DIM + k0 + 32, 0, 3);
    v16h_t b0 = load_frag_b(wq_h, DIM, col0,      k0);
    v16h_t b1 = load_frag_b(wq_h, DIM, col0 + 16, k0);
    acc0 = wmma_f16(a, b0, acc0);
    acc1 = wmma_f16(a, b1, acc1);
  }
  const int rbase = row0 + ((l >> 4) << 3);
#pragma unroll
  for (int r = 0; r < 8; ++r) {
    int row = rbase + r;
    int n = row >> 11, m = row & (M_SEQ - 1);
#pragma unroll
    for (int t = 0; t < 2; ++t) {
      int col = col0 + t * 16 + (l & 15);
      float v = (t == 0 ? acc0[r] : acc1[r]) + qkv_b[col];
      int which = col >> 9;            // 0:q 1:k 2:v
      int cc = col & (DIM - 1);
      int h = cc >> 6, d = cc & (HDIM - 1);
      size_t o = ((size_t)(n * HEADS + h) * M_SEQ + m) * HDIM + d;
      if (which == 0)      { v *= SCALE; q_f[o] = v; q_h[o] = (_Float16)v; }
      else if (which == 1) { k_h[o] = (_Float16)v; }
      else                 { v_h[o] = (_Float16)v; }
    }
  }
}

// ---------------------------------------------------------------------------
// Kernel 4: fused router — logits via WMMA + streaming top-16 per query row.
// Key tiles staged global->LDS once per WG (async DMA, double-buffered, the
// DMA for tile t+1 overlaps WMMA + selection of tile t); B fragments come
// from LDS; q fragments live in registers for the whole 2048-key sweep.
// Never materializes the 536 MB logits tensor.
// ---------------------------------------------------------------------------
__global__ __launch_bounds__(256)
void router_topk_kernel(const _Float16* __restrict__ q_h,
                        const _Float16* __restrict__ k_h,
                        int* __restrict__ idx) {
  __shared__ float    sc[128 * 16];          // 8 KB score staging
  __shared__ _Float16 kbuf[2][16 * HDIM];    // 2 x 2 KB key-tile double buffer
  const int bh   = blockIdx.y;
  const int row0 = blockIdx.x * 128;
  const int w    = threadIdx.x >> 5;
  const int l    = lane_id();
  const _Float16* qb = q_h + (size_t)bh * M_SEQ * HDIM;
  const _Float16* kb = k_h + (size_t)bh * M_SEQ * HDIM;

  v16h_t a0 = load_frag_a(qb, HDIM, row0 + w * 16, 0);
  v16h_t a1 = load_frag_a(qb, HDIM, row0 + w * 16, 32);

  float vals[TOPK]; int ids[TOPK];
#pragma unroll
  for (int j = 0; j < TOPK; ++j) { vals[j] = -3.4e38f; ids[j] = j; }
  float vmin = -3.4e38f; int amin = 0;

  const int NT = M_SEQ / 16;                 // 128 key tiles
  stage_tile(kb, kbuf[0], 0);                // prologue: tile 0
  wait_async_zero();
  __syncthreads();

  for (int kt = 0; kt < NT; ++kt) {
    const int cur = kt & 1;
    if (kt + 1 < NT) stage_tile(kb, kbuf[cur ^ 1], (kt + 1) * 16);

    v16h_t b0 = load_frag_b_lds(kbuf[cur], 0);
    v16h_t b1 = load_frag_b_lds(kbuf[cur], 32);
    v8f_t c = {};
    c = wmma_f16(a0, b0, c);
    c = wmma_f16(a1, b1, c);

    __syncthreads();                         // prior selection done reading sc
    const int rb = w * 16 + ((l >> 4) << 3);
#pragma unroll
    for (int r = 0; r < 8; ++r) sc[(rb + r) * 16 + (l & 15)] = c[r];
    __syncthreads();                         // sc visible

    if (threadIdx.x < 128) {
      const int key0 = kt * 16;
#pragma unroll
      for (int s = 0; s < 16; ++s) {
        float v = sc[threadIdx.x * 16 + s];
        if (v > vmin) {
          vals[amin] = v; ids[amin] = key0 + s;
          vmin = vals[0]; amin = 0;
#pragma unroll
          for (int j = 1; j < TOPK; ++j)
            if (vals[j] < vmin) { vmin = vals[j]; amin = j; }
        }
      }
    }
    wait_async_zero();                       // next tile's DMA landed
    __syncthreads();
  }

  if (threadIdx.x < 128) {
    int* o = idx + ((size_t)bh * M_SEQ + row0 + threadIdx.x) * TOPK;
#pragma unroll
    for (int j = 0; j < TOPK; ++j) o[j] = ids[j];
  }
}

// ---------------------------------------------------------------------------
// Kernel 5: gathered 16-key attention. One wave32 per query; lane owns two
// of the 64 dims; dots via __shfl_xor butterfly; f32 softmax; merged-head f16.
// ---------------------------------------------------------------------------
__global__ __launch_bounds__(256)
void attn_kernel(const float* __restrict__ q_f,
                 const _Float16* __restrict__ k_h,
                 const _Float16* __restrict__ v_h,
                 const int* __restrict__ idx,
                 _Float16* __restrict__ attn_h) {
  const int wq = blockIdx.x * 8 + (threadIdx.x >> 5);   // query over BH*M
  const int l  = lane_id();
  const int bh = wq >> 11;
  const int m  = wq & (M_SEQ - 1);
  const int n  = bh >> 3, h = bh & 7;

  const float2 q2 = *(const float2*)(q_f + ((size_t)bh * M_SEQ + m) * HDIM + l * 2);
  const int* ip = idx + ((size_t)bh * M_SEQ + m) * TOPK;

  float s[TOPK]; int kid[TOPK];
  float smax = -3.4e38f;
#pragma unroll
  for (int j = 0; j < TOPK; ++j) {
    kid[j] = ip[j];
    const _Float16* kp = k_h + ((size_t)bh * M_SEQ + kid[j]) * HDIM + l * 2;
    float p = q2.x * (float)kp[0] + q2.y * (float)kp[1];
#pragma unroll
    for (int off = 16; off > 0; off >>= 1) p += __shfl_xor(p, off, 32);
    s[j] = p;
    smax = fmaxf(smax, p);
  }
  float sum = 0.f;
#pragma unroll
  for (int j = 0; j < TOPK; ++j) { s[j] = __expf(s[j] - smax); sum += s[j]; }
  const float inv = 1.f / sum;
  float ax = 0.f, ay = 0.f;
#pragma unroll
  for (int j = 0; j < TOPK; ++j) {
    const _Float16* vp = v_h + ((size_t)bh * M_SEQ + kid[j]) * HDIM + l * 2;
    float wgt = s[j] * inv;
    ax += wgt * (float)vp[0];
    ay += wgt * (float)vp[1];
  }
  _Float16* op = attn_h + ((size_t)n * M_SEQ + m) * DIM + h * HDIM + l * 2;
  op[0] = (_Float16)ax;
  op[1] = (_Float16)ay;
}

// ---------------------------------------------------------------------------
// Kernel 6: output GEMM  out(8192x512) = attn(8192x512) * Wo^T + b  (f32 out)
// ---------------------------------------------------------------------------
__global__ __launch_bounds__(256)
void out_gemm_kernel(const _Float16* __restrict__ attn_h,
                     const _Float16* __restrict__ wo_h,
                     const float* __restrict__ wo_b,
                     float* __restrict__ out) {
  const int w = threadIdx.x >> 5;
  const int l = lane_id();
  const int row0 = blockIdx.y * 128 + w * 16;
  const int col0 = blockIdx.x * 32;
  v8f_t acc0 = {}, acc1 = {};
  for (int k0 = 0; k0 < DIM; k0 += 32) {
    v16h_t a  = load_frag_a(attn_h, DIM, row0, k0);
    if (k0 + 32 < DIM)
      __builtin_prefetch(attn_h + (size_t)(row0 + (l & 15)) * DIM + k0 + 32, 0, 3);
    v16h_t b0 = load_frag_b(wo_h, DIM, col0,      k0);
    v16h_t b1 = load_frag_b(wo_h, DIM, col0 + 16, k0);
    acc0 = wmma_f16(a, b0, acc0);
    acc1 = wmma_f16(a, b1, acc1);
  }
  const int rbase = row0 + ((l >> 4) << 3);
  const float b0v = wo_b[col0 + (l & 15)];
  const float b1v = wo_b[col0 + 16 + (l & 15)];
#pragma unroll
  for (int r = 0; r < 8; ++r) {
    int row = rbase + r;
    out[(size_t)row * DIM + col0 + (l & 15)]      = acc0[r] + b0v;
    out[(size_t)row * DIM + col0 + 16 + (l & 15)] = acc1[r] + b1v;
  }
}

// ---------------------------------------------------------------------------
// Host-side launcher
// ---------------------------------------------------------------------------
extern "C" void kernel_launch(void* const* d_in, const int* in_sizes, int n_in,
                              void* d_out, int out_size, void* d_ws, size_t ws_size,
                              hipStream_t stream) {
  (void)in_sizes; (void)n_in; (void)out_size; (void)ws_size;
  const float* x     = (const float*)d_in[0];
  const float* pos_w = (const float*)d_in[1];
  const float* pos_b = (const float*)d_in[2];
  const float* qkv_w = (const float*)d_in[3];
  const float* qkv_b = (const float*)d_in[4];
  const float* wo_w  = (const float*)d_in[5];
  const float* wo_b  = (const float*)d_in[6];
  float* out = (float*)d_out;

  // Workspace carve-out (256 B aligned); total ~62 MB.
  char* p = (char*)d_ws;
  auto carve = [&](size_t bytes) -> char* {
    char* r = p;
    p += (bytes + 255) & ~(size_t)255;
    return r;
  };
  _Float16* xh     = (_Float16*)carve((size_t)ROWS * DIM * 2);
  _Float16* wq_h   = (_Float16*)carve((size_t)QKV_N * DIM * 2);
  _Float16* wo_h   = (_Float16*)carve((size_t)DIM * DIM * 2);
  float*    q_f    = (float*)   carve((size_t)BH * M_SEQ * HDIM * 4);
  _Float16* q_h    = (_Float16*)carve((size_t)BH * M_SEQ * HDIM * 2);
  _Float16* k_h    = (_Float16*)carve((size_t)BH * M_SEQ * HDIM * 2);
  _Float16* v_h    = (_Float16*)carve((size_t)BH * M_SEQ * HDIM * 2);
  int*      idx    = (int*)     carve((size_t)BH * M_SEQ * TOPK * 4);
  _Float16* attn_h = (_Float16*)carve((size_t)ROWS * DIM * 2);

  conv_pos_kernel<<<(ROWS * DIM) / 256, 256, 0, stream>>>(x, pos_w, pos_b, xh);
  cvt_weights_kernel<<<(QKV_N * DIM + DIM * DIM) / 256, 256, 0, stream>>>(
      qkv_w, wo_w, wq_h, wo_h);
  qkv_gemm_kernel<<<dim3(QKV_N / 32, ROWS / 128), 256, 0, stream>>>(
      xh, wq_h, qkv_b, q_f, q_h, k_h, v_h);
  router_topk_kernel<<<dim3(M_SEQ / 128, BH), 256, 0, stream>>>(q_h, k_h, idx);
  attn_kernel<<<(BH * M_SEQ) / 8, 256, 0, stream>>>(q_f, k_h, v_h, idx, attn_h);
  out_gemm_kernel<<<dim3(DIM / 32, ROWS / 128), 256, 0, stream>>>(
      attn_h, wo_h, wo_b, out);
}
